// Attention_82136954568859
// MI455X (gfx1250) — matrix operand order repaired and checked by reading, not compiled, not run on hardware
//
#include <hip/hip_runtime.h>
#include <hip/hip_bf16.h>

// ---------------------------------------------------------------------------
// MI455X (gfx1250) attention block, compile-only target.
// Matmuls via v_wmma_f32_16x16x32_bf16 (wave32 WMMA); GEMM tile staging via
// the Tensor Data Mover (tensor_load_to_lds + s_wait_tensorcnt) when the
// toolchain exposes the builtin, else plain global->LDS vector loads.
// ---------------------------------------------------------------------------

#define DEV __device__ __forceinline__

typedef __attribute__((ext_vector_type(16))) __bf16          v16bf;
typedef __attribute__((ext_vector_type(8)))  float           v8f;
typedef __attribute__((ext_vector_type(8)))  unsigned short  u16x8;
typedef __attribute__((ext_vector_type(4)))  unsigned int    u32x4;
typedef __attribute__((ext_vector_type(8)))  int             i32x8;
typedef __attribute__((ext_vector_type(4)))  int             i32x4;

union Frag16 { v16bf bf; u16x8 h[2]; };

constexpr int S_LEN   = 2048;
constexpr int D_MODEL = 4096;
constexpr int NHEAD   = 32;
constexpr int NKVH    = 8;
constexpr int HEADDIM = 128;
constexpr int NQCOLS  = NHEAD * HEADDIM;   // 4096
constexpr int NKVCOLS = NKVH  * HEADDIM;   // 1024
constexpr float NEG_MASK = -1.0e9f;

#if defined(__gfx1250__) && __has_builtin(__builtin_amdgcn_tensor_load_to_lds) && \
    __has_builtin(__builtin_amdgcn_s_wait_tensorcnt)
#define ATHENA_USE_TDM 1
#else
#define ATHENA_USE_TDM 0
#endif

DEV unsigned short f2bf(float f) {
  union { float f; unsigned u; } c; c.f = f;
  unsigned r = c.u + 0x7fffu + ((c.u >> 16) & 1u);   // round-to-nearest-even
  return (unsigned short)(r >> 16);
}
DEV float bf2f(unsigned short h) {
  union { unsigned u; float f; } c; c.u = ((unsigned)h) << 16;
  return c.f;
}

#if ATHENA_USE_TDM
// ---------------------------------------------------------------------------
// TDM: DMA a (rows up to tile_dim1) x 32-element bf16 tile (row stride = K
// elements) from global memory into LDS at byte offset lds_off.
// D# packing per CDNA5 ISA 8.3/8.4 (2D tensor, groups 2/3 zero/NULL-like).
// ---------------------------------------------------------------------------
DEV void tdm_load_tile_bf16(unsigned lds_off, const unsigned short* gtile,
                            unsigned tensor_d0, unsigned tensor_d1,
                            unsigned tile_d0, unsigned tile_d1) {
  unsigned long long ga = (unsigned long long)(const void*)gtile;
  u32x4 g0;
  g0[0] = 1u;                                            // count=1, user mode
  g0[1] = lds_off;                                       // lds_addr (bytes)
  g0[2] = (unsigned)(ga & 0xffffffffull);                // global_addr[31:0]
  g0[3] = (unsigned)((ga >> 32) & 0x1ffffffull)          // global_addr[56:32]
        | (2u << 30);                                    // type = 2 ("image")
  const unsigned st0 = tensor_d0;                        // dim0 stride = K
  i32x8 g1;
  g1[0] = (int)(1u << 16);                               // data_size = 2 bytes
  g1[1] = (int)((tensor_d0 & 0xffffu) << 16);            // tensor_dim0[15:0]
  g1[2] = (int)((tensor_d0 >> 16) | ((tensor_d1 & 0xffffu) << 16));
  g1[3] = (int)((tensor_d1 >> 16) | (tile_d0 << 16));    // tile_dim0
  g1[4] = (int)(tile_d1);                                // tile_dim1, tile_dim2=0
  g1[5] = (int)st0;                                      // dim0_stride[31:0]
  g1[6] = 0;                                             // stride hi, dim1_stride lo
  g1[7] = 0;
  i32x4 z4 = {0, 0, 0, 0};
#if defined(__clang_major__) && (__clang_major__ >= 23)
  i32x8 z8 = {0, 0, 0, 0, 0, 0, 0, 0};
  __builtin_amdgcn_tensor_load_to_lds(g0, g1, z4, z4, z8, 0);
#else
  __builtin_amdgcn_tensor_load_to_lds(g0, g1, z4, z4, 0);
#endif
}
#endif  // ATHENA_USE_TDM

// ---------------------------------------------------------------------------
// f32 -> bf16 conversion (grid-stride)
// ---------------------------------------------------------------------------
__global__ void convert_f32_bf16(const float* __restrict__ in,
                                 unsigned short* __restrict__ out, long n) {
  long stride = (long)gridDim.x * blockDim.x;
  for (long i = (long)blockIdx.x * blockDim.x + threadIdx.x; i < n; i += stride)
    out[i] = f2bf(in[i]);
}

// ---------------------------------------------------------------------------
// RoPE in-place on bf16 (s, nheads*128) buffer. Pairs (2i, 2i+1) within head.
// ---------------------------------------------------------------------------
__global__ void rope_bf16(unsigned short* __restrict__ t,
                          const float* __restrict__ cosb,
                          const float* __restrict__ sinb, int nheads) {
  const int HALF = HEADDIM / 2;  // 64
  long total = (long)S_LEN * nheads * HALF;
  long stride = (long)gridDim.x * blockDim.x;
  for (long idx = (long)blockIdx.x * blockDim.x + threadIdx.x; idx < total; idx += stride) {
    int i = (int)(idx % HALF);
    long tmp = idx / HALF;
    int h = (int)(tmp % nheads);
    int s = (int)(tmp / nheads);
    long base = ((long)s * nheads + h) * HEADDIM + 2 * i;
    float tr = bf2f(t[base]);
    float ti = bf2f(t[base + 1]);
    float c  = cosb[s * HALF + i];
    float sn = sinb[s * HALF + i];
    t[base]     = f2bf(tr * c - ti * sn);
    t[base + 1] = f2bf(tr * sn + ti * c);
  }
}

// ---------------------------------------------------------------------------
// C(MxN) = A(MxK) @ W(NxK)^T, bf16 inputs, f32 accumulate via WMMA.
// Block tile 128x128, BK=32. 8 waves: wave (wm 0..3, wn 0..1) owns 32x64.
// Tile staging: TDM DMA (wave 0 issues 2 descriptors, waits TENSORcnt) when
// available; otherwise cooperative b128 global loads.
// Fragment loads from LDS follow the documented CDNA5 layouts:
//   A 16x32: lanes 0-15 row M=l, elems 0..7 = K[h*8..], 8..15 = K[16+h*8..]
//   B 32x16: lanes 0-15 col N=l, elems 0..7 = K[h*8..], 8..15 = K[16+h*8..]
// ---------------------------------------------------------------------------
template <typename OutT>
DEV void store_out(OutT* C, long idx, float v);
template <> DEV void store_out<float>(float* C, long idx, float v) { C[idx] = v; }
template <> DEV void store_out<unsigned short>(unsigned short* C, long idx, float v) { C[idx] = f2bf(v); }

template <typename OutT>
__global__ __launch_bounds__(256)
void gemm_bf16_wmma(const unsigned short* __restrict__ A,
                    const unsigned short* __restrict__ W,
                    OutT* __restrict__ C,
                    int M, int N, int K) {
  constexpr int BM = 128, BN = 128, BK = 32;
  __shared__ __attribute__((aligned(16))) unsigned short sA[BM][BK];
  __shared__ __attribute__((aligned(16))) unsigned short sW[BN][BK];

  const int tid  = threadIdx.x;
  const int lane = tid & 31;
  const int wv   = tid >> 5;
  const int wm   = (wv & 3) * 32;
  const int wn   = (wv >> 2) * 64;
  const int hlf  = lane >> 4;
  const int l16  = lane & 15;
  const long blockM = (long)blockIdx.y * BM;
  const long blockN = (long)blockIdx.x * BN;

  const v8f vzero = {0.f, 0.f, 0.f, 0.f, 0.f, 0.f, 0.f, 0.f};
  v8f acc[2][4];
#pragma unroll
  for (int i = 0; i < 2; ++i)
#pragma unroll
    for (int j = 0; j < 4; ++j) acc[i][j] = vzero;

  const int ldRow = tid >> 1;
  const int ldCol = (tid & 1) * 16;

  for (int k0 = 0; k0 < K; k0 += BK) {
#if ATHENA_USE_TDM
    if (wv == 0) {
      tdm_load_tile_bf16((unsigned)(size_t)(void*)&sA[0][0],
                         A + blockM * (long)K + k0,
                         (unsigned)K, (unsigned)M, BK, BM);
      tdm_load_tile_bf16((unsigned)(size_t)(void*)&sW[0][0],
                         W + blockN * (long)K + k0,
                         (unsigned)K, (unsigned)N, BK, BN);
      __builtin_amdgcn_s_wait_tensorcnt(0);
    }
    __syncthreads();
#else
    const unsigned short* gA = A + (blockM + ldRow) * (long)K + k0 + ldCol;
    const unsigned short* gW = W + (blockN + ldRow) * (long)K + k0 + ldCol;
    *(u16x8*)&sA[ldRow][ldCol]     = *(const u16x8*)gA;
    *(u16x8*)&sA[ldRow][ldCol + 8] = *(const u16x8*)(gA + 8);
    *(u16x8*)&sW[ldRow][ldCol]     = *(const u16x8*)gW;
    *(u16x8*)&sW[ldRow][ldCol + 8] = *(const u16x8*)(gW + 8);
    __builtin_prefetch(gA + BK, 0, 0);   // global_prefetch_b8, next k-tile
    __builtin_prefetch(gW + BK, 0, 0);
    __syncthreads();
#endif

    Frag16 afrag[2], bfrag[4];
#pragma unroll
    for (int i = 0; i < 2; ++i) {
      const int m = wm + i * 16 + l16;
      afrag[i].h[0] = *(const u16x8*)&sA[m][hlf * 8];
      afrag[i].h[1] = *(const u16x8*)&sA[m][16 + hlf * 8];
    }
#pragma unroll
    for (int j = 0; j < 4; ++j) {
      const int n = wn + j * 16 + l16;
      bfrag[j].h[0] = *(const u16x8*)&sW[n][hlf * 8];
      bfrag[j].h[1] = *(const u16x8*)&sW[n][16 + hlf * 8];
    }
#pragma unroll
    for (int i = 0; i < 2; ++i)
#pragma unroll
      for (int j = 0; j < 4; ++j)
        acc[i][j] = __builtin_amdgcn_wmma_f32_16x16x32_bf16(
            false, afrag[i].bf, false, bfrag[j].bf, (short)0, acc[i][j],
            false, false);
    __syncthreads();
  }

  // C/D layout: VGPR r, lanes 0-15 -> M=r, lanes 16-31 -> M=r+8, N=lane&15
#pragma unroll
  for (int i = 0; i < 2; ++i)
#pragma unroll
    for (int j = 0; j < 4; ++j)
#pragma unroll
      for (int r = 0; r < 8; ++r) {
        const long row = blockM + wm + i * 16 + r + hlf * 8;
        const long col = blockN + wn + j * 16 + l16;
        store_out<OutT>(C, row * N + col, acc[i][j][r]);
      }
}

// ---------------------------------------------------------------------------
// Flash-style causal GQA attention. Block = 64 q-rows x 1 head, 4 waves.
// Each wave: 16 q-rows. KV tiles of 32. K stays row-major in LDS;
// V is stored transposed so P@V B-fragments are contiguous b128 LDS reads.
// P (f32 C/D layout) is relaid to bf16 A layout via per-wave LDS scratch.
// ---------------------------------------------------------------------------
__global__ __launch_bounds__(128)
void attn_wmma(const unsigned short* __restrict__ Q,
               const unsigned short* __restrict__ Kb,
               const unsigned short* __restrict__ Vb,
               unsigned short* __restrict__ O) {
  constexpr int QT = 64;
  constexpr int KT = 32;
  __shared__ __attribute__((aligned(16))) unsigned short sK[KT][HEADDIM];
  __shared__ __attribute__((aligned(16))) unsigned short sVt[HEADDIM][KT];
  __shared__ __attribute__((aligned(16))) unsigned short sP[4][16][KT];

  const int head  = blockIdx.y;
  const int kvh   = head / (NHEAD / NKVH);
  const int qbase = blockIdx.x * QT;
  const int tid = threadIdx.x, lane = tid & 31, wv = tid >> 5;
  const int hlf = lane >> 4, l16 = lane & 15;
  const int qrow0 = qbase + wv * 16;

  // Q fragments for this wave's 16 rows: 4 chunks of K=32 over HD=128
  Frag16 qfrag[4];
  const unsigned short* qp = Q + (long)(qrow0 + l16) * NQCOLS + head * HEADDIM;
#pragma unroll
  for (int c = 0; c < 4; ++c) {
    qfrag[c].h[0] = *(const u16x8*)(qp + c * 32 + hlf * 8);
    qfrag[c].h[1] = *(const u16x8*)(qp + c * 32 + 16 + hlf * 8);
  }

  const v8f vzero = {0.f, 0.f, 0.f, 0.f, 0.f, 0.f, 0.f, 0.f};
  v8f o_acc[8];
#pragma unroll
  for (int j = 0; j < 8; ++j) o_acc[j] = vzero;
  float m_run[8], l_run[8];
#pragma unroll
  for (int r = 0; r < 8; ++r) { m_run[r] = -3.0e38f; l_run[r] = 0.f; }

  const float scale = 0.088388347648318447f;  // 1/sqrt(128)

  for (int kv0 = 0; kv0 < qbase + QT; kv0 += KT) {
    // Cooperative load: K row-major, V transposed. 32x128 tile, 128 threads.
#pragma unroll
    for (int p = 0; p < 4; ++p) {
      const int idx = tid + p * 128;
      const int row = idx >> 4;
      const int cs  = (idx & 15) * 8;
      const long g = (long)(kv0 + row) * NKVCOLS + kvh * HEADDIM + cs;
      *(u16x8*)&sK[row][cs] = *(const u16x8*)(Kb + g);
      u16x8 vvv = *(const u16x8*)(Vb + g);
#pragma unroll
      for (int e = 0; e < 8; ++e) sVt[cs + e][row] = vvv[e];
    }
    __syncthreads();

    // S = Q @ K^T : two 16x16 N-tiles, K-chain of 4 WMMAs each
    v8f s_acc[2];
    s_acc[0] = vzero; s_acc[1] = vzero;
#pragma unroll
    for (int nt = 0; nt < 2; ++nt)
#pragma unroll
      for (int c = 0; c < 4; ++c) {
        Frag16 kf;
        kf.h[0] = *(const u16x8*)&sK[nt * 16 + l16][c * 32 + hlf * 8];
        kf.h[1] = *(const u16x8*)&sK[nt * 16 + l16][c * 32 + 16 + hlf * 8];
        s_acc[nt] = __builtin_amdgcn_wmma_f32_16x16x32_bf16(
            false, qfrag[c].bf, false, kf.bf, (short)0, s_acc[nt], false, false);
      }

    // Causal mask + online softmax (rows live across 16-lane halves)
    float sv0[8], sv1[8], corr[8];
#pragma unroll
    for (int r = 0; r < 8; ++r) {
      const int qg = qrow0 + r + hlf * 8;
      const int kg0 = kv0 + l16;
      const int kg1 = kv0 + 16 + l16;
      sv0[r] = s_acc[0][r] * scale + (kg0 <= qg ? 0.f : NEG_MASK);
      sv1[r] = s_acc[1][r] * scale + (kg1 <= qg ? 0.f : NEG_MASK);
      float mx = fmaxf(sv0[r], sv1[r]);
#pragma unroll
      for (int m = 1; m <= 8; m <<= 1) mx = fmaxf(mx, __shfl_xor(mx, m, 32));
      const float mn = fmaxf(m_run[r], mx);
      corr[r] = __expf(m_run[r] - mn);
      m_run[r] = mn;
      const float p0 = __expf(sv0[r] - mn);
      const float p1 = __expf(sv1[r] - mn);
      sP[wv][r + hlf * 8][l16]      = f2bf(p0);
      sP[wv][r + hlf * 8][16 + l16] = f2bf(p1);
      float s = p0 + p1;
#pragma unroll
      for (int m = 1; m <= 8; m <<= 1) s += __shfl_xor(s, m, 32);
      l_run[r] = l_run[r] * corr[r] + s;
    }
#pragma unroll
    for (int j = 0; j < 8; ++j)
#pragma unroll
      for (int r = 0; r < 8; ++r) o_acc[j][r] = o_acc[j][r] * corr[r];

    // O += P @ V (wave-local sP write->read; DS is in-order per wave)
    Frag16 pf;
    pf.h[0] = *(const u16x8*)&sP[wv][l16][hlf * 8];
    pf.h[1] = *(const u16x8*)&sP[wv][l16][16 + hlf * 8];
#pragma unroll
    for (int j = 0; j < 8; ++j) {
      Frag16 vf;
      vf.h[0] = *(const u16x8*)&sVt[j * 16 + l16][hlf * 8];
      vf.h[1] = *(const u16x8*)&sVt[j * 16 + l16][16 + hlf * 8];
      o_acc[j] = __builtin_amdgcn_wmma_f32_16x16x32_bf16(
          false, pf.bf, false, vf.bf, (short)0, o_acc[j], false, false);
    }
    __syncthreads();
  }

  // Normalize and store attention output (bf16, feeds final projection)
#pragma unroll
  for (int j = 0; j < 8; ++j)
#pragma unroll
    for (int r = 0; r < 8; ++r) {
      const long row = qrow0 + r + hlf * 8;
      const long col = head * HEADDIM + j * 16 + l16;
      O[row * NQCOLS + col] = f2bf(o_acc[j][r] / l_run[r]);
    }
}

// ---------------------------------------------------------------------------
// Host-side orchestration
// ---------------------------------------------------------------------------
extern "C" void kernel_launch(void* const* d_in, const int* in_sizes, int n_in,
                              void* d_out, int out_size, void* d_ws, size_t ws_size,
                              hipStream_t stream) {
  (void)in_sizes; (void)n_in; (void)out_size; (void)ws_size;
  const float* x    = (const float*)d_in[0];
  const float* wq   = (const float*)d_in[1];
  const float* wk   = (const float*)d_in[2];
  const float* wv   = (const float*)d_in[3];
  const float* wo   = (const float*)d_in[4];
  const float* fcos = (const float*)d_in[5];
  const float* fsin = (const float*)d_in[6];
  float* out = (float*)d_out;

  size_t off = 0;
  auto take = [&](size_t elems) {
    unsigned short* p = (unsigned short*)((char*)d_ws + off);
    off += ((elems * sizeof(unsigned short) + 255) & ~(size_t)255);
    return p;
  };
  unsigned short* xb  = take((size_t)S_LEN * D_MODEL);
  unsigned short* wqb = take((size_t)NQCOLS * D_MODEL);
  unsigned short* wkb = take((size_t)NKVCOLS * D_MODEL);
  unsigned short* wvb = take((size_t)NKVCOLS * D_MODEL);
  unsigned short* wob = take((size_t)D_MODEL * NQCOLS);
  unsigned short* qb  = take((size_t)S_LEN * NQCOLS);
  unsigned short* kb  = take((size_t)S_LEN * NKVCOLS);
  unsigned short* vb  = take((size_t)S_LEN * NKVCOLS);
  unsigned short* ob  = take((size_t)S_LEN * NQCOLS);

  auto cv = [&](const float* in, unsigned short* o, long n) {
    convert_f32_bf16<<<2048, 256, 0, stream>>>(in, o, n);
  };
  cv(x,  xb,  (long)S_LEN * D_MODEL);
  cv(wq, wqb, (long)NQCOLS * D_MODEL);
  cv(wk, wkb, (long)NKVCOLS * D_MODEL);
  cv(wv, wvb, (long)NKVCOLS * D_MODEL);
  cv(wo, wob, (long)D_MODEL * NQCOLS);

  // Q/K/V projections
  gemm_bf16_wmma<unsigned short><<<dim3(NQCOLS / 128, S_LEN / 128), 256, 0, stream>>>(
      xb, wqb, qb, S_LEN, NQCOLS, D_MODEL);
  gemm_bf16_wmma<unsigned short><<<dim3(NKVCOLS / 128, S_LEN / 128), 256, 0, stream>>>(
      xb, wkb, kb, S_LEN, NKVCOLS, D_MODEL);
  gemm_bf16_wmma<unsigned short><<<dim3(NKVCOLS / 128, S_LEN / 128), 256, 0, stream>>>(
      xb, wvb, vb, S_LEN, NKVCOLS, D_MODEL);

  // RoPE
  rope_bf16<<<2048, 256, 0, stream>>>(qb, fcos, fsin, NHEAD);
  rope_bf16<<<2048, 256, 0, stream>>>(kb, fcos, fsin, NKVH);

  // Causal GQA attention
  attn_wmma<<<dim3(S_LEN / 64, NHEAD), 128, 0, stream>>>(qb, kb, vb, ob);

  // Output projection -> f32 d_out
  gemm_bf16_wmma<float><<<dim3(D_MODEL / 128, S_LEN / 128), 256, 0, stream>>>(
      ob, wob, out, S_LEN, D_MODEL, NQCOLS);
}
